// ScaledDotProductAttention_90271622627864
// MI455X (gfx1250) — matrix operand order repaired and checked by reading, not compiled, not run on hardware
//
#include <hip/hip_runtime.h>
#include <cstdint>

typedef __attribute__((ext_vector_type(16))) _Float16 v16h;
typedef __attribute__((ext_vector_type(8)))  _Float16 v8h;
typedef __attribute__((ext_vector_type(8)))  float    v8f;

#define S_LEN 2048
#define D_DIM 64
#define BH    32            // B*H fused
#define BK    64            // keys staged per LDS tile
#define QW    32            // query rows per wave = two 16-row M tiles
#define WAVES 4             // 4 x wave32 = 128 threads
#define QB    (QW * WAVES)  // 128 query rows per block
#define KS    72            // K LDS row stride in halves (64 + 8 pad)
#define VS    72            // Vt LDS row stride in halves
#define PS    72            // P LDS row stride in halves (64 keys + 8 pad)

#define KH_BYTES ((size_t)BH * S_LEN * D_DIM * 2)   // 8 MB per f16 tensor

static __device__ __forceinline__ uint32_t lds_off(const void* p) {
    // generic -> LDS address space; ptrtoint of AS(3) gives the raw LDS offset
    return (uint32_t)(uintptr_t)(__attribute__((address_space(3))) const void*)p;
}

static __device__ __forceinline__ v16h load16(const _Float16* p) {
    union { v16h v; v8h h[2]; } u;
    u.h[0] = *(const v8h*)(p);
    u.h[1] = *(const v8h*)(p + 8);
    return u.v;
}
static __device__ __forceinline__ v16h load16_2(const _Float16* p0, const _Float16* p1) {
    union { v16h v; v8h h[2]; } u;
    u.h[0] = *(const v8h*)(p0);
    u.h[1] = *(const v8h*)(p1);
    return u.v;
}

// ---- pre-pass: K -> f16 [bh][s][d]; V -> f16 transposed [bh][d][s] ----
__global__ __launch_bounds__(256)
void cvt_kernel(const float* __restrict__ K, const float* __restrict__ V,
                _Float16* __restrict__ Kh, _Float16* __restrict__ Vt)
{
    const size_t i = (size_t)blockIdx.x * 256 + threadIdx.x;
    Kh[i] = (_Float16)K[i];                       // coalesced both sides
    const size_t bh  = i / (S_LEN * D_DIM);       // output-indexed transpose:
    const size_t rem = i % (S_LEN * D_DIM);       // coalesced f16 writes
    const size_t d   = rem / S_LEN;
    const size_t s   = rem % S_LEN;
    Vt[i] = (_Float16)V[(bh * S_LEN + s) * D_DIM + d];
}

// ---- flash-attention forward: f16 WMMA, f32 accumulate, online softmax ----
template <bool ASYNC>
__global__ __launch_bounds__(128)
void fa_kernel(const float* __restrict__ Kf, const float* __restrict__ Qf,
               const float* __restrict__ Vf, float* __restrict__ O,
               const _Float16* __restrict__ Kh, const _Float16* __restrict__ Vth)
{
    __shared__ __align__(16) _Float16 Kt[2][BK * KS];     // [key][d]  f16, double buffered
    __shared__ __align__(16) _Float16 Vt[2][D_DIM * VS];  // [d][key]  f16, double buffered
    __shared__ __align__(16) _Float16 Pt[WAVES * QW * PS];// per-wave P relayout scratch

    const int tid  = threadIdx.x;
    const int lane = tid & 31;
    const int wave = tid >> 5;
    const int bh   = blockIdx.y;
    const int q0w  = blockIdx.x * QB + wave * QW;

    const int r  = lane & 15;
    const int hi = lane >> 4;

    const size_t headoff = (size_t)bh * S_LEN * D_DIM;
    const _Float16* kh = Kh  + headoff;           // f16 [s][d]
    const _Float16* vh = Vth + headoff;           // f16 [d][s]

    // ---- Q: two 16x64 tiles -> four 16x32 f16 A-fragments (registers) ----
    // ISA 7.12.2 A-layout: element e -> K = (e<8 ? hi*8+e : 16+hi*8+(e-8))
    v16h qa[2][2];
#pragma unroll
    for (int qt = 0; qt < 2; ++qt) {
        const float* qrow = Qf + ((size_t)bh * S_LEN + q0w + qt * 16 + r) * D_DIM;
#pragma unroll
        for (int g = 0; g < 2; ++g) {
            v16h a;
#pragma unroll
            for (int e = 0; e < 16; ++e) {
                const int kk = (e < 8) ? (hi * 8 + e) : (16 + hi * 8 + (e - 8));
                a[e] = (_Float16)qrow[g * 32 + kk];
            }
            qa[qt][g] = a;
        }
    }

    v8f o[2][4];                                  // 2 M tiles x 4 d tiles
#pragma unroll
    for (int qt = 0; qt < 2; ++qt)
#pragma unroll
        for (int t = 0; t < 4; ++t) o[qt][t] = (v8f){};
    float mrow[2][8], lrow[2][8];
#pragma unroll
    for (int qt = 0; qt < 2; ++qt)
#pragma unroll
        for (int j = 0; j < 8; ++j) { mrow[qt][j] = -__builtin_inff(); lrow[qt][j] = 0.0f; }

    _Float16* pmine = &Pt[wave * QW * PS];
    const float scale = 0.125f;                   // 1/sqrt(64)

    // ---- tile staging ----
    auto stage = [&](int buf, int kb) {
        if (ASYNC) {
            // async L2->LDS copies (8 per wave per tile), tracked by ASYNCcnt
#pragma unroll
            for (int i = 0; i < 4; ++i) {
                const int chunk = i * 128 + tid;  // 512 x 16B chunks per tensor
                const int row = chunk >> 3;       // K: key row ; V: d row
                const int c   = chunk & 7;
                {
                    const _Float16* gbase = kh + (size_t)kb * D_DIM;
                    const int      goff   = row * (D_DIM * 2) + c * 16;
                    const uint32_t loff   = lds_off(&Kt[buf][row * KS]) + c * 16;
                    asm volatile("global_load_async_to_lds_b128 %0, %1, %2"
                                 :: "v"(loff), "v"(goff), "s"(gbase) : "memory");
                }
                {
                    const _Float16* gbase = vh;
                    const int      goff   = row * (S_LEN * 2) + kb * 2 + c * 16;
                    const uint32_t loff   = lds_off(&Vt[buf][row * VS]) + c * 16;
                    asm volatile("global_load_async_to_lds_b128 %0, %1, %2"
                                 :: "v"(loff), "v"(goff), "s"(gbase) : "memory");
                }
            }
        } else {
            // fallback: synchronous load + convert + LDS store
#pragma unroll
            for (int i = 0; i < (BK * D_DIM) / 128; ++i) {
                const int idx = i * 128 + tid;
                const int kr = idx >> 6, dc = idx & 63;
                Kt[buf][kr * KS + dc] = (_Float16)Kf[headoff + (size_t)(kb + kr) * D_DIM + dc];
                Vt[buf][dc * VS + kr] = (_Float16)Vf[headoff + (size_t)(kb + kr) * D_DIM + dc];
            }
        }
    };

    stage(0, 0);

    for (int kb = 0; kb < S_LEN; kb += BK) {
        const int cur = (kb / BK) & 1;
        const bool has_next = (kb + BK) < S_LEN;
        if (has_next) stage(cur ^ 1, kb + BK);    // overlap next tile's copies
        if (ASYNC) {
            if (has_next) asm volatile("s_wait_asynccnt 0x8" ::: "memory");
            else          asm volatile("s_wait_asynccnt 0x0" ::: "memory");
        } else if (has_next) {
            __builtin_prefetch(&Kf[headoff + (size_t)(kb + BK) * D_DIM + tid * 16], 0, 1);
        }
        __syncthreads();

        const _Float16* Kc = &Kt[cur][0];
        const _Float16* Vc = &Vt[cur][0];

        // ---- scores: 64 keys x 2 M tiles, B-fragments shared across M ----
        v8f s[2][4];
#pragma unroll
        for (int c = 0; c < 4; ++c) {
            const _Float16* krow = Kc + (c * 16 + r) * KS;
            const v16h bl  = load16_2(krow + hi * 8, krow + 16 + hi * 8);        // d = 0..31
            const v16h bh2 = load16_2(krow + 32 + hi * 8, krow + 48 + hi * 8);   // d = 32..63
#pragma unroll
            for (int qt = 0; qt < 2; ++qt) {
                v8f acc = {};
                acc = __builtin_amdgcn_wmma_f32_16x16x32_f16(false, qa[qt][0], false, bl,  (short)0, acc, false, false);
                acc = __builtin_amdgcn_wmma_f32_16x16x32_f16(false, qa[qt][1], false, bh2, (short)0, acc, false, false);
                s[qt][c] = acc;
            }
        }

        // ---- online softmax over the 64-key group (one reduction pass) ----
#pragma unroll
        for (int qt = 0; qt < 2; ++qt) {
#pragma unroll
            for (int j = 0; j < 8; ++j) {
                const float x0 = s[qt][0][j] * scale;
                const float x1 = s[qt][1][j] * scale;
                const float x2 = s[qt][2][j] * scale;
                const float x3 = s[qt][3][j] * scale;
                float mx = fmaxf(fmaxf(x0, x1), fmaxf(x2, x3));
#pragma unroll
                for (int off = 1; off < 16; off <<= 1)
                    mx = fmaxf(mx, __shfl_xor(mx, off, 32));
                const float mnew  = fmaxf(mrow[qt][j], mx);
                const float alpha = __expf(mrow[qt][j] - mnew);
                const float p0 = __expf(x0 - mnew);
                const float p1 = __expf(x1 - mnew);
                const float p2 = __expf(x2 - mnew);
                const float p3 = __expf(x3 - mnew);
                float rs = (p0 + p1) + (p2 + p3);
#pragma unroll
                for (int off = 1; off < 16; off <<= 1)
                    rs += __shfl_xor(rs, off, 32);
                lrow[qt][j] = lrow[qt][j] * alpha + rs;
                mrow[qt][j] = mnew;
#pragma unroll
                for (int t = 0; t < 4; ++t) o[qt][t][j] *= alpha;
                // store P in C-layout: row = qt*16 + hi*8 + j, cols c*16 + r
                _Float16* pr = pmine + (qt * 16 + hi * 8 + j) * PS;
                pr[r]      = (_Float16)p0;
                pr[16 + r] = (_Float16)p1;
                pr[32 + r] = (_Float16)p2;
                pr[48 + r] = (_Float16)p3;
            }
        }

        // ---- re-layout P -> A-fragments (wave-private LDS; in-order per wave) ----
        v16h pa[2][2];
#pragma unroll
        for (int qt = 0; qt < 2; ++qt)
#pragma unroll
            for (int kf = 0; kf < 2; ++kf) {
                const _Float16* pr = pmine + (qt * 16 + r) * PS + kf * 32;
                pa[qt][kf] = load16_2(pr + hi * 8, pr + 16 + hi * 8);
            }

        // ---- O += P * V : V B-fragments shared across both M tiles ----
#pragma unroll
        for (int t = 0; t < 4; ++t) {
#pragma unroll
            for (int kf = 0; kf < 2; ++kf) {
                const v16h vb = load16_2(Vc + (t * 16 + r) * VS + kf * 32 + hi * 8, Vc + (t * 16 + r) * VS + kf * 32 + 16 + hi * 8);
#pragma unroll
                for (int qt = 0; qt < 2; ++qt)
                    o[qt][t] = __builtin_amdgcn_wmma_f32_16x16x32_f16(false, pa[qt][kf], false, vb, (short)0, o[qt][t], false, false);
            }
        }
        __syncthreads();   // all waves done with buf[cur]; it may be overwritten
    }

    // ---- normalize and write out ----
#pragma unroll
    for (int qt = 0; qt < 2; ++qt)
#pragma unroll
        for (int j = 0; j < 8; ++j) {
            const int m = qt * 16 + hi * 8 + j;
            const float inv = 1.0f / lrow[qt][j];
            const size_t rowoff = ((size_t)bh * S_LEN + q0w + m) * D_DIM;
            O[rowoff + 0 * 16 + r] = o[qt][0][j] * inv;
            O[rowoff + 1 * 16 + r] = o[qt][1][j] * inv;
            O[rowoff + 2 * 16 + r] = o[qt][2][j] * inv;
            O[rowoff + 3 * 16 + r] = o[qt][3][j] * inv;
        }
}

extern "C" void kernel_launch(void* const* d_in, const int* in_sizes, int n_in,
                              void* d_out, int out_size, void* d_ws, size_t ws_size,
                              hipStream_t stream) {
    (void)in_sizes; (void)n_in; (void)out_size;
    // setup_inputs() dict order: k, q, v
    const float* k = (const float*)d_in[0];
    const float* q = (const float*)d_in[1];
    const float* v = (const float*)d_in[2];
    float* o = (float*)d_out;

    const dim3 grid(S_LEN / QB, BH);
    if (ws_size >= 2 * KH_BYTES) {
        _Float16* Kh = (_Float16*)d_ws;
        _Float16* Vt = (_Float16*)((char*)d_ws + KH_BYTES);
        const size_t total = (size_t)BH * S_LEN * D_DIM;
        cvt_kernel<<<(unsigned)(total / 256), 256, 0, stream>>>(k, v, Kh, Vt);
        fa_kernel<true><<<grid, 128, 0, stream>>>(k, q, v, o, Kh, Vt);
    } else {
        fa_kernel<false><<<grid, 128, 0, stream>>>(k, q, v, o, nullptr, nullptr);
    }
}
